// EvoformerODEFunc_55860344652389
// MI455X (gfx1250) — matrix operand order repaired and checked by reading, not compiled, not run on hardware
//
#include <hip/hip_runtime.h>

// ---------------------------------------------------------------------------
// Types & helpers
// ---------------------------------------------------------------------------
typedef __bf16 bf16;
typedef __attribute__((ext_vector_type(16))) __bf16 v16bf;
typedef __attribute__((ext_vector_type(8)))  __bf16 bf16x8;
typedef __attribute__((ext_vector_type(4)))  __bf16 bf16x4;
typedef __attribute__((ext_vector_type(8)))  float  v8f;
typedef __attribute__((ext_vector_type(4)))  float  f32x4;

__device__ inline unsigned short f2bf_bits(float f) {
  union { float f; unsigned int u; } v; v.f = f;
  unsigned int u = v.u;
  unsigned int r = (u + 0x7FFFu + ((u >> 16) & 1u)) >> 16;  // RNE
  return (unsigned short)r;
}
__device__ inline bf16 f2bf(float f) {
  union { unsigned short s; bf16 h; } c; c.s = f2bf_bits(f); return c.h;
}
__device__ inline float bf2f(bf16 h) {
  union { unsigned short s; bf16 h; } c; c.h = h;
  union { unsigned int u; float f; } v; v.u = ((unsigned int)c.s) << 16;
  return v.f;
}
__device__ inline float sigf(float x) { return 1.0f / (1.0f + __expf(-x)); }

// A-operand K permutation: position of K k in the fragment-ordered layout is
// pk32(k) (swap bits 3 and 4). Lane (lg) then reads contiguous [lg*16, +16).
__device__ inline int pk32(int k) {
  return (k & 7) | ((k & 8) << 1) | ((k & 16) >> 1);
}

// Load a 16-element bf16 fragment from 32 contiguous bytes (16B aligned).
__device__ inline v16bf ldfrag(const bf16* p) {
  union { struct { bf16x8 lo, hi; } h; v16bf v; } f;
  f.h.lo = *(const bf16x8*)p;
  f.h.hi = *(const bf16x8*)(p + 8);
  return f.v;
}

__device__ inline v8f wmma_bf(v16bf a, v16bf b, v8f c) {
  return __builtin_amdgcn_wmma_f32_16x16x32_bf16(false, a, false, b, (short)0, c,
                                                 false, false);
}

enum { EPI_NONE = 0, EPI_RELU = 1, EPI_SIG = 2, EPI_SIGMUL = 3, EPI_ADDRES = 4,
       EPI_MULADD = 5 };

__device__ inline void stg(float* p, float v) { *p = v; }
__device__ inline void stg(bf16* p, float v)  { *p = f2bf(v); }

// ---------------------------------------------------------------------------
// Generic batched WMMA GEMM:  C = epi(A @ B + bias)
//   A: bf16 [M x K] row-major (or [K x M] if TA), B: bf16 [K x N] row-major.
//   LDS: A tile in K-permuted layout (frag-contiguous), B tile transposed.
//   Requires M%64==0, K%32==0, N%8==0.
// ---------------------------------------------------------------------------
template <int EPI, bool TA, typename OUT_T>
__global__ __launch_bounds__(256) void gemm_k(
    const bf16* __restrict__ A, const bf16* __restrict__ B,
    const float* __restrict__ bias, OUT_T* __restrict__ C,
    const float* __restrict__ aux, const float* __restrict__ res,
    int M, int N, int K, long long sA, long long sB, long long sC) {
  __shared__ bf16 As[64][40];   // [m][pk32(k)]
  __shared__ bf16 Bs[64][40];   // [n][pk32(k)]   (transposed B tile)
  const int t = threadIdx.x;
  const int lane = t & 31;
  const int wv = t >> 5;
  const int lg = lane >> 4, lr = lane & 15;
  const int mi = wv & 3, half = wv >> 2;
  const int m0b = blockIdx.y * 64, n0b = blockIdx.x * 64;
  const long long z = blockIdx.z;
  const bf16* Ab = A + z * sA;
  const bf16* Bb = B + z * sB;

  v8f acc0 = {}, acc1 = {};

  for (int k0 = 0; k0 < K; k0 += 32) {
    if (TA) {  // A stored [K x M]
      int kk = t >> 3, mm = (t & 7) * 8;
      bf16x8 vv = *(const bf16x8*)(Ab + (long long)(k0 + kk) * M + (m0b + mm));
      int pc = pk32(kk);
#pragma unroll
      for (int j = 0; j < 8; ++j) As[mm + j][pc] = vv[j];
    } else {
      int mm = t >> 2, kk = (t & 3) * 8;
      bf16x8 vv = *(const bf16x8*)(Ab + (long long)(m0b + mm) * K + (k0 + kk));
      *(bf16x8*)&As[mm][pk32(kk)] = vv;   // group of 8 stays contiguous
    }
    {
      int kk = t >> 3, nn = (t & 7) * 8;
      int gn = n0b + nn;
      if (gn < N) {
        bf16x8 vv = *(const bf16x8*)(Bb + (long long)(k0 + kk) * N + gn);
#pragma unroll
        for (int j = 0; j < 8; ++j) Bs[nn + j][pk32(kk)] = vv[j];
      } else {
#pragma unroll
        for (int j = 0; j < 8; ++j) Bs[nn + j][pk32(kk)] = f2bf(0.0f);
      }
    }
    __syncthreads();
    v16bf af = ldfrag(&As[mi * 16 + lr][lg * 16]);
    v16bf b0 = ldfrag(&Bs[half * 32 + lr][lg * 16]);
    v16bf b1 = ldfrag(&Bs[half * 32 + 16 + lr][lg * 16]);
    acc0 = wmma_bf(af, b0, acc0);
    acc1 = wmma_bf(af, b1, acc1);
    __syncthreads();
  }

  OUT_T* Cb = C + z * sC;
#pragma unroll
  for (int v = 0; v < 8; ++v) {
    int gm = m0b + mi * 16 + v + 8 * lg;
#pragma unroll
    for (int nt = 0; nt < 2; ++nt) {
      int gn = n0b + half * 32 + nt * 16 + lr;
      if (gn < N) {
        float x = (nt == 0) ? acc0[v] : acc1[v];
        if (bias) x += bias[gn];
        long long idx = (long long)gm * N + gn;
        if (EPI == EPI_RELU)        x = x > 0.0f ? x : 0.0f;
        else if (EPI == EPI_SIG)    x = sigf(x);
        else if (EPI == EPI_SIGMUL) x = sigf(x) * aux[idx];
        else if (EPI == EPI_ADDRES) x = res[idx] + x;
        else if (EPI == EPI_MULADD) x = res[idx] + aux[idx] * x;
        stg(&Cb[idx], x);
      }
    }
  }
}

// ---------------------------------------------------------------------------
// LayerNorm over last dim D (D%32==0, D<=256), one wave per row, bf16 out.
// ---------------------------------------------------------------------------
__global__ __launch_bounds__(256) void ln_k(const float* __restrict__ x,
                                            const float* __restrict__ g,
                                            const float* __restrict__ b,
                                            bf16* __restrict__ out, int D) {
  int lane = threadIdx.x & 31, wv = threadIdx.x >> 5;
  long long row = (long long)blockIdx.x * 8 + wv;
  const float* xr = x + row * D;
  int n = D >> 5;
  float vals[8];
  float s = 0.0f;
#pragma unroll
  for (int i = 0; i < 8; ++i)
    if (i < n) { vals[i] = xr[lane + i * 32]; s += vals[i]; }
#pragma unroll
  for (int m = 1; m < 32; m <<= 1) s += __shfl_xor(s, m, 32);
  float mu = s / (float)D;
  float vs = 0.0f;
#pragma unroll
  for (int i = 0; i < 8; ++i)
    if (i < n) { float d = vals[i] - mu; vs += d * d; }
#pragma unroll
  for (int m = 1; m < 32; m <<= 1) vs += __shfl_xor(vs, m, 32);
  float rstd = rsqrtf(vs / (float)D + 1e-5f);
#pragma unroll
  for (int i = 0; i < 8; ++i)
    if (i < n) {
      int c = lane + i * 32;
      out[row * D + c] = f2bf((vals[i] - mu) * rstd * g[c] + b[c]);
    }
}

// LayerNorm (no affine) of tu stored d-major [64][RR] -> bf16 [RR][64].
__global__ __launch_bounds__(256) void ln_tu_k(const float* __restrict__ tu,
                                               bf16* __restrict__ out,
                                               long long RR) {
  long long pos = (long long)blockIdx.x * blockDim.x + threadIdx.x;
  float v[64];
  float s = 0.0f;
#pragma unroll
  for (int d = 0; d < 64; ++d) { v[d] = tu[(long long)d * RR + pos]; s += v[d]; }
  float mu = s * (1.0f / 64.0f);
  float vs = 0.0f;
#pragma unroll
  for (int d = 0; d < 64; ++d) { float dd = v[d] - mu; vs += dd * dd; }
  float rstd = rsqrtf(vs * (1.0f / 64.0f) + 1e-5f);
#pragma unroll
  for (int d = 0; d < 64; ++d) out[pos * 64 + d] = f2bf((v[d] - mu) * rstd);
}

// Transpose f32 [R x C] -> bf16 [C x R] with scale. R%32==0, C%32==0.
__global__ __launch_bounds__(256) void transpose_cvt_k(
    const float* __restrict__ in, bf16* __restrict__ out, int R, int C,
    float scale) {
  __shared__ float T[32][33];
  int r0 = blockIdx.y * 32, c0 = blockIdx.x * 32;
  int tx = threadIdx.x & 31, ty = threadIdx.x >> 5;
#pragma unroll
  for (int j = 0; j < 32; j += 8)
    T[ty + j][tx] = in[(long long)(r0 + ty + j) * C + (c0 + tx)];
  __syncthreads();
#pragma unroll
  for (int j = 0; j < 32; j += 8)
    out[(long long)(c0 + ty + j) * R + (r0 + tx)] = f2bf(T[tx][ty + j] * scale);
}

// Pair bias: biasT[h][row] = zb[row,:] . W[:,h] + b[h]   (row = q*384 + k)
__global__ __launch_bounds__(128) void pair_bias_k(
    const bf16* __restrict__ zb, const float* __restrict__ w,
    const float* __restrict__ b, float* __restrict__ biasT, long long RR) {
  __shared__ float red[4][4];
  long long row = blockIdx.x;
  int t = threadIdx.x, lane = t & 31, wv = t >> 5;
  float xv = bf2f(zb[row * 128 + t]);
  float a0 = xv * w[t * 4 + 0];
  float a1 = xv * w[t * 4 + 1];
  float a2 = xv * w[t * 4 + 2];
  float a3 = xv * w[t * 4 + 3];
#pragma unroll
  for (int m = 1; m < 32; m <<= 1) {
    a0 += __shfl_xor(a0, m, 32); a1 += __shfl_xor(a1, m, 32);
    a2 += __shfl_xor(a2, m, 32); a3 += __shfl_xor(a3, m, 32);
  }
  if (lane == 0) { red[wv][0] = a0; red[wv][1] = a1; red[wv][2] = a2; red[wv][3] = a3; }
  __syncthreads();
  if (t < 4) {
    float s = red[0][t] + red[1][t] + red[2][t] + red[3][t] + b[t];
    biasT[(long long)t * RR + row] = s;
  }
}

// ---------------------------------------------------------------------------
// MSA row attention (flash, per (s, head, 128-row q block)).
//   qkv: f32 [s*384][192] (q|k|v each 64 wide, head h at offset h*16).
//   biasT: f32 [4][384][384]. attn out: f32 [s*384][64].
// ---------------------------------------------------------------------------
__global__ __launch_bounds__(256) void row_attn_k(
    const float* __restrict__ qkv, const float* __restrict__ biasT,
    float* __restrict__ attn) {
  __shared__ bf16 Qs[128][40];       // [qrow][pk32(k)], K padded to 32
  __shared__ bf16 Ks[64][40];        // [keycol][pk32(headdim k)] (transposed, padded)
  __shared__ bf16 Vs[16][72];        // [headdim n][perm(keycol k)]
  __shared__ bf16 Ps[8][16][72];     // per-wave P tile, perm layout over 64 keys
  const int t = threadIdx.x, lane = t & 31, wv = t >> 5;
  const int lg = lane >> 4, lr = lane & 15;
  const int qb = blockIdx.x;   // 0..2
  const int h  = blockIdx.y;   // 0..3
  const int s  = blockIdx.z;   // 0..63
  const long long base = (long long)s * 384;
  const long long RRc = 147456;

  {
    int row = t >> 1, c8 = (t & 1) * 8;
    const float* src = qkv + (base + qb * 128 + row) * 192 + h * 16 + c8;
    f32x4 q0 = *(const f32x4*)src;
    f32x4 q1 = *(const f32x4*)(src + 4);
    bf16x8 qv;
#pragma unroll
    for (int j = 0; j < 4; ++j) { qv[j] = f2bf(q0[j]); qv[4 + j] = f2bf(q1[j]); }
    *(bf16x8*)&Qs[row][pk32(c8)] = qv;
    bf16x8 zv;
#pragma unroll
    for (int j = 0; j < 8; ++j) zv[j] = f2bf(0.0f);
    *(bf16x8*)&Qs[row][pk32(16 + c8)] = zv;
  }
  v8f acc = {};
  float mrow[8], lrow[8];
#pragma unroll
  for (int v = 0; v < 8; ++v) { mrow[v] = -1e30f; lrow[v] = 0.0f; }
  __syncthreads();

  for (int kb = 0; kb < 6; ++kb) {
    {
      int kc = t >> 2, kd4 = (t & 3) * 4;
      f32x4 kv = *(const f32x4*)(qkv + (base + kb * 64 + kc) * 192 + 64 + h * 16 + kd4);
      bf16x4 kb4;
#pragma unroll
      for (int j = 0; j < 4; ++j) kb4[j] = f2bf(kv[j]);
      *(bf16x4*)&Ks[kc][pk32(kd4)] = kb4;
      f32x4 vv = *(const f32x4*)(qkv + (base + kb * 64 + kc) * 192 + 128 + h * 16 + kd4);
#pragma unroll
      for (int j = 0; j < 4; ++j) Vs[kd4 + j][(kc & 32) + pk32(kc & 31)] = f2bf(vv[j]);
      if (t < 64) {
        bf16x8 zv;
#pragma unroll
        for (int j = 0; j < 8; ++j) zv[j] = f2bf(0.0f);
        *(bf16x8*)&Ks[t][8] = zv;
        *(bf16x8*)&Ks[t][24] = zv;
      }
    }
    __syncthreads();

    // S = Q K^T (K contraction 16, zero padded to 32)
    v16bf qf = ldfrag(&Qs[wv * 16 + lr][lg * 16]);
    v8f st[4];
#pragma unroll
    for (int nt = 0; nt < 4; ++nt) {
      v16bf bf_ = ldfrag(&Ks[nt * 16 + lr][lg * 16]);
      v8f zz = {};
      st[nt] = wmma_bf(qf, bf_, zz);
    }
    // online softmax over this 64-col block (+ pair bias)
#pragma unroll
    for (int v = 0; v < 8; ++v) {
      int qrow = qb * 128 + wv * 16 + v + 8 * lg;
      float sv[4];
      float mx = -1e30f;
#pragma unroll
      for (int nt = 0; nt < 4; ++nt) {
        int col = kb * 64 + nt * 16 + lr;
        float x = st[nt][v] * 0.25f +
                  biasT[(long long)h * RRc + (long long)qrow * 384 + col];
        sv[nt] = x;
        mx = fmaxf(mx, x);
      }
#pragma unroll
      for (int m = 1; m < 16; m <<= 1) mx = fmaxf(mx, __shfl_xor(mx, m, 32));
      float nm = fmaxf(mrow[v], mx);
      float sf = __expf(mrow[v] - nm);
      float psum = 0.0f;
#pragma unroll
      for (int nt = 0; nt < 4; ++nt) {
        float p = __expf(sv[nt] - nm);
        psum += p;
        int k = nt * 16 + lr;                   // key index within 64
        int pos = (k & 32) + pk32(k & 31);      // perm layout position
        Ps[wv][v + 8 * lg][pos] = f2bf(p);
      }
#pragma unroll
      for (int m = 1; m < 16; m <<= 1) psum += __shfl_xor(psum, m, 32);
      lrow[v] = lrow[v] * sf + psum;
      mrow[v] = nm;
      acc[v] *= sf;
    }
    __syncthreads();
    // acc += P @ V  (K = 64 keys -> 2 WMMA steps)
    v16bf a0 = ldfrag(&Ps[wv][lr][lg * 16]);
    v16bf a1 = ldfrag(&Ps[wv][lr][32 + lg * 16]);
    v16bf vb0 = ldfrag(&Vs[lr][lg * 16]);
    v16bf vb1 = ldfrag(&Vs[lr][32 + lg * 16]);
    acc = wmma_bf(a0, vb0, acc);
    acc = wmma_bf(a1, vb1, acc);
    __syncthreads();
  }
#pragma unroll
  for (int v = 0; v < 8; ++v) {
    int qrow = qb * 128 + wv * 16 + v + 8 * lg;
    attn[(base + qrow) * 64 + h * 16 + lr] = acc[v] / lrow[v];
  }
}

// ---------------------------------------------------------------------------
// MSA column attention: per residue r, X = mp[:, r, :] (64x64),
//   P = softmax(X X^T / 8), out = P X  -> f32 [s*384][64].
// ---------------------------------------------------------------------------
__global__ __launch_bounds__(256) void col_attn_k(const bf16* __restrict__ mp,
                                                  float* __restrict__ out) {
  __shared__ bf16 Xs[64][72];    // [sq][perm(h)]  (same as XsA, B for scores)
  __shared__ bf16 XsA[64][72];   // [sq][perm(h)]  (A-operand layout)
  __shared__ bf16 XsT[64][72];   // [h][perm(sq)]        (B for mo)
  __shared__ float Ss[64][68];
  __shared__ bf16 Ps[64][72];    // [sq][perm(st)] (A-operand layout)
  const int t = threadIdx.x, lane = t & 31, wv = t >> 5;
  const int lg = lane >> 4, lr = lane & 15;
  const int ri = blockIdx.x;
  const int mi = wv & 3, half = wv >> 2;
  {
    int sq = t >> 2, c = (t & 3) * 16;
    const bf16* src = mp + ((long long)(sq * 384 + ri)) * 64 + c;
    bf16x8 lo = *(const bf16x8*)src;
    bf16x8 hi = *(const bf16x8*)(src + 8);
    *(bf16x8*)&Xs[sq][(c & 32) + pk32(c & 31)] = lo;
    *(bf16x8*)&Xs[sq][(c & 32) + pk32((c & 31) + 8)] = hi;
    int cb = c & 32;
    *(bf16x8*)&XsA[sq][cb + pk32(c & 31)] = lo;
    *(bf16x8*)&XsA[sq][cb + pk32((c & 31) + 8)] = hi;
#pragma unroll
    for (int j = 0; j < 8; ++j) { XsT[c + j][(sq & 32) + pk32(sq & 31)] = lo[j]; XsT[c + 8 + j][(sq & 32) + pk32(sq & 31)] = hi[j]; }
  }
  __syncthreads();
  // scores = X X^T / 8
#pragma unroll
  for (int nt = 0; nt < 2; ++nt) {
    v8f acc = {};
#pragma unroll
    for (int k0 = 0; k0 < 64; k0 += 32) {
      v16bf af = ldfrag(&XsA[mi * 16 + lr][k0 + lg * 16]);
      v16bf bf_ = ldfrag(&Xs[half * 32 + nt * 16 + lr][k0 + lg * 16]);
      acc = wmma_bf(af, bf_, acc);
    }
#pragma unroll
    for (int v = 0; v < 8; ++v)
      Ss[mi * 16 + v + 8 * lg][half * 32 + nt * 16 + lr] = acc[v] * 0.125f;
  }
  __syncthreads();
  // softmax rows (write P in A-operand perm layout)
#pragma unroll
  for (int i = 0; i < 8; ++i) {
    int row = wv * 8 + i;
    float x0 = Ss[row][lane], x1 = Ss[row][lane + 32];
    float mx = fmaxf(x0, x1);
#pragma unroll
    for (int m = 1; m < 32; m <<= 1) mx = fmaxf(mx, __shfl_xor(mx, m, 32));
    float e0 = __expf(x0 - mx), e1 = __expf(x1 - mx);
    float sm = e0 + e1;
#pragma unroll
    for (int m = 1; m < 32; m <<= 1) sm += __shfl_xor(sm, m, 32);
    float inv = 1.0f / sm;
    int c0 = lane, c1 = lane + 32;
    Ps[row][(c0 & 32) + pk32(c0 & 31)] = f2bf(e0 * inv);
    Ps[row][(c1 & 32) + pk32(c1 & 31)] = f2bf(e1 * inv);
  }
  __syncthreads();
  // mo = P @ X
#pragma unroll
  for (int nt = 0; nt < 2; ++nt) {
    v8f acc = {};
#pragma unroll
    for (int k0 = 0; k0 < 64; k0 += 32) {
      v16bf af = ldfrag(&Ps[mi * 16 + lr][k0 + lg * 16]);
      v16bf bf_ = ldfrag(&XsT[half * 32 + nt * 16 + lr][k0 + lg * 16]);
      acc = wmma_bf(af, bf_, acc);
    }
#pragma unroll
    for (int v = 0; v < 8; ++v) {
      int sq = mi * 16 + v + 8 * lg;
      int hc = half * 32 + nt * 16 + lr;
      out[((long long)(sq * 384 + ri)) * 64 + hc] = acc[v];
    }
  }
}

// ---------------------------------------------------------------------------
// Small kernels
// ---------------------------------------------------------------------------
__global__ void cvt_k(const float* __restrict__ in, bf16* __restrict__ out,
                      long long n) {
  long long i = (long long)blockIdx.x * blockDim.x + threadIdx.x;
  if (i < n) out[i] = f2bf(in[i]);
}

__global__ void time_k(const float* __restrict__ t, const float* __restrict__ w1,
                       const float* __restrict__ b1, const float* __restrict__ w2,
                       const float* __restrict__ b2, float* __restrict__ mix) {
  __shared__ float r0s[2], r1s[2];
  int j = threadIdx.x;  // 64 threads
  float h = t[0] * w1[j] + b1[j];
  float sl = h * sigf(h);
  float p0 = sl * w2[j * 2 + 0];
  float p1 = sl * w2[j * 2 + 1];
#pragma unroll
  for (int m = 1; m < 32; m <<= 1) {
    p0 += __shfl_xor(p0, m, 32);
    p1 += __shfl_xor(p1, m, 32);
  }
  int lane = j & 31, wv = j >> 5;
  if (lane == 0) { r0s[wv] = p0; r1s[wv] = p1; }
  __syncthreads();
  if (j == 0) mix[0] = sigf(r0s[0] + r0s[1] + b2[0]);
  if (j == 1) mix[1] = sigf(r1s[0] + r1s[1] + b2[1]);
}

__global__ void final_k(const float* __restrict__ m3, const float* __restrict__ m0,
                        const float* __restrict__ z3, const float* __restrict__ z0,
                        const float* __restrict__ mix, float* __restrict__ out,
                        long long MT, long long ZT) {
  long long i = (long long)blockIdx.x * blockDim.x + threadIdx.x;
  if (i < MT)            out[i] = (m3[i] - m0[i]) * mix[0];
  else if (i < MT + ZT) { long long j = i - MT; out[MT + j] = (z3[j] - z0[j]) * mix[1]; }
}

// ---------------------------------------------------------------------------
// Host orchestration
// ---------------------------------------------------------------------------
#define LAUNCH_GEMM(EPI, TA, OT, A, B, BIAS, C, AUX, RES, M, N, K, SA, SB, SC, BATCH) \
  gemm_k<EPI, TA, OT><<<dim3((unsigned)(((N) + 63) / 64),                             \
                            (unsigned)(((M) + 63) / 64), (unsigned)(BATCH)),          \
                       256, 0, stream>>>((A), (B), (BIAS), (C), (AUX), (RES), (M),    \
                                         (N), (K), (long long)(SA),                   \
                                         (long long)(SB), (long long)(SC))

extern "C" void kernel_launch(void* const* d_in, const int* in_sizes, int n_in,
                              void* d_out, int out_size, void* d_ws, size_t ws_size,
                              hipStream_t stream) {
  (void)in_sizes; (void)n_in; (void)out_size; (void)ws_size;
  const int S = 64, R = 384;
  const long long SR = (long long)S * R;          // 24576
  const long long RR = (long long)R * R;          // 147456
  const long long MT = SR * 256;                  // m elements
  const long long ZT = RR * 128;                  // z elements

  const float* m0 = (const float*)d_in[0];
  const float* z0 = (const float*)d_in[1];
  const float* t_in = (const float*)d_in[2];
  int ii = 3;
#define NEXTP ((const float*)d_in[ii++])
  const float* rn_g = NEXTP;   const float* rn_b = NEXTP;
  const float* qkv_w = NEXTP;  const float* qkv_b = NEXTP;
  const float* rgate_w = NEXTP;const float* rgate_b = NEXTP;
  const float* rout_w = NEXTP; const float* rout_b = NEXTP;
  const float* pbn_g = NEXTP;  const float* pbn_b = NEXTP;
  const float* pbp_w = NEXTP;  const float* pbp_b = NEXTP;
  const float* cn_g = NEXTP;   const float* cn_b = NEXTP;
  const float* cproj_w = NEXTP;const float* cproj_b = NEXTP;
  const float* cgate_w = NEXTP;const float* cgate_b = NEXTP;
  const float* cout_w = NEXTP; const float* cout_b = NEXTP;
  const float* mtn_g = NEXTP;  const float* mtn_b = NEXTP;
  const float* mt1_w = NEXTP;  const float* mt1_b = NEXTP;
  const float* mt2_w = NEXTP;  const float* mt2_b = NEXTP;
  const float* on_g = NEXTP;   const float* on_b = NEXTP;
  const float* opa_w = NEXTP;  const float* opa_b = NEXTP;
  const float* opb_w = NEXTP;  const float* opb_b = NEXTP;
  const float* oout_w = NEXTP; const float* oout_b = NEXTP;
  const float* trn_g = NEXTP;  const float* trn_b = NEXTP;
  const float* tpa_w = NEXTP;  const float* tpa_b = NEXTP;
  const float* tpb_w = NEXTP;  const float* tpb_b = NEXTP;
  const float* tgw = NEXTP;    const float* tgb = NEXTP;
  const float* tout_w = NEXTP; const float* tout_b = NEXTP;
  const float* ptn_g = NEXTP;  const float* ptn_b = NEXTP;
  const float* pt1_w = NEXTP;  const float* pt1_b = NEXTP;
  const float* pt2_w = NEXTP;  const float* pt2_b = NEXTP;
  const float* tm1_w = NEXTP;  const float* tm1_b = NEXTP;
  const float* tm2_w = NEXTP;  const float* tm2_b = NEXTP;
#undef NEXTP

  size_t off = 0;
  auto alloc = [&](size_t bytes) -> void* {
    void* p = (char*)d_ws + off;
    off = (off + bytes + 255) & ~(size_t)255;
    return p;
  };
  auto cvt = [&](const float* src, bf16* dst, long long n) {
    cvt_k<<<(unsigned)((n + 255) / 256), 256, 0, stream>>>(src, dst, n);
  };

  // bf16 weights
  bf16* Wqkv = (bf16*)alloc(49152 * 2);   cvt(qkv_w, Wqkv, 49152);
  bf16* Wrg  = (bf16*)alloc(16384 * 2);   cvt(rgate_w, Wrg, 16384);
  bf16* Wro  = (bf16*)alloc(16384 * 2);   cvt(rout_w, Wro, 16384);
  bf16* Wcp  = (bf16*)alloc(16384 * 2);   cvt(cproj_w, Wcp, 16384);
  bf16* Wcg  = (bf16*)alloc(16384 * 2);   cvt(cgate_w, Wcg, 16384);
  bf16* Wco  = (bf16*)alloc(16384 * 2);   cvt(cout_w, Wco, 16384);
  bf16* Wm1  = (bf16*)alloc(262144 * 2);  cvt(mt1_w, Wm1, 262144);
  bf16* Wm2  = (bf16*)alloc(262144 * 2);  cvt(mt2_w, Wm2, 262144);
  bf16* Wpa  = (bf16*)alloc(8192 * 2);    cvt(opa_w, Wpa, 8192);
  bf16* Wpb  = (bf16*)alloc(8192 * 2);    cvt(opb_w, Wpb, 8192);
  bf16* Woo  = (bf16*)alloc(4096 * 2);    cvt(oout_w, Woo, 4096);
  bf16* Wta  = (bf16*)alloc(8192 * 2);    cvt(tpa_w, Wta, 8192);
  bf16* Wtb  = (bf16*)alloc(8192 * 2);    cvt(tpb_w, Wtb, 8192);
  bf16* Wtg  = (bf16*)alloc(16384 * 2);   cvt(tgw, Wtg, 16384);
  bf16* Wto  = (bf16*)alloc(8192 * 2);    cvt(tout_w, Wto, 8192);
  bf16* Wp1  = (bf16*)alloc(65536 * 2);   cvt(pt1_w, Wp1, 65536);
  bf16* Wp2  = (bf16*)alloc(65536 * 2);   cvt(pt2_w, Wp2, 65536);

  // scratch
  bf16*  mn_bf   = (bf16*)alloc((size_t)SR * 256 * 2);
  float* qkv_f   = (float*)alloc((size_t)SR * 192 * 4);
  bf16*  zn_bf   = (bf16*)alloc((size_t)RR * 128 * 2);
  float* biasT   = (float*)alloc((size_t)4 * RR * 4);
  float* attn_f  = (float*)alloc((size_t)SR * 64 * 4);
  bf16*  gated   = (bf16*)alloc((size_t)SR * 64 * 2);
  float* m_a     = (float*)alloc((size_t)SR * 256 * 4);
  float* m_b     = (float*)alloc((size_t)SR * 256 * 4);
  bf16*  mp_bf   = (bf16*)alloc((size_t)SR * 64 * 2);
  bf16*  hid_m   = (bf16*)alloc((size_t)SR * 1024 * 2);
  float* a_f     = (float*)alloc((size_t)SR * 32 * 4);
  float* b_f     = (float*)alloc((size_t)SR * 32 * 4);
  bf16*  a_t     = (bf16*)alloc((size_t)SR * 32 * 2);
  bf16*  b_t     = (bf16*)alloc((size_t)SR * 32 * 2);
  float* om_f    = (float*)alloc((size_t)32 * RR * 4);
  bf16*  om_t    = (bf16*)alloc((size_t)RR * 32 * 2);
  float* ta_f    = (float*)alloc((size_t)RR * 64 * 4);
  float* tb_f    = (float*)alloc((size_t)RR * 64 * 4);
  float* tg_f    = (float*)alloc((size_t)RR * 128 * 4);
  bf16*  ta_t    = (bf16*)alloc((size_t)RR * 64 * 2);
  bf16*  tb_t    = (bf16*)alloc((size_t)RR * 64 * 2);
  float* tu_f    = (float*)alloc((size_t)64 * RR * 4);
  bf16*  tu_bf   = (bf16*)alloc((size_t)RR * 64 * 2);
  bf16*  hid_z   = (bf16*)alloc((size_t)RR * 512 * 2);
  float* z_a     = (float*)alloc((size_t)RR * 128 * 4);
  float* z_b     = (float*)alloc((size_t)RR * 128 * 4);
  float* mix     = (float*)alloc(2 * 4);
  float* colat_f = attn_f;  // reuse (stage A consumer done before stage B)

  // ---- Stage A: MSA row attention with pair bias ----
  ln_k<<<(unsigned)(SR / 8), 256, 0, stream>>>(m0, rn_g, rn_b, mn_bf, 256);
  LAUNCH_GEMM(EPI_NONE, false, float, mn_bf, Wqkv, qkv_b, qkv_f, nullptr, nullptr,
              (int)SR, 192, 256, 0, 0, 0, 1);
  ln_k<<<(unsigned)(RR / 8), 256, 0, stream>>>(z0, pbn_g, pbn_b, zn_bf, 128);
  pair_bias_k<<<(unsigned)RR, 128, 0, stream>>>(zn_bf, pbp_w, pbp_b, biasT, RR);
  row_attn_k<<<dim3(3, 4, 64), 256, 0, stream>>>(qkv_f, biasT, attn_f);
  LAUNCH_GEMM(EPI_SIGMUL, false, bf16, mn_bf, Wrg, rgate_b, gated, attn_f, nullptr,
              (int)SR, 64, 256, 0, 0, 0, 1);
  LAUNCH_GEMM(EPI_ADDRES, false, float, gated, Wro, rout_b, m_a, nullptr, m0,
              (int)SR, 256, 64, 0, 0, 0, 1);

  // ---- Stage B: MSA column attention ----
  ln_k<<<(unsigned)(SR / 8), 256, 0, stream>>>(m_a, cn_g, cn_b, mn_bf, 256);
  LAUNCH_GEMM(EPI_NONE, false, bf16, mn_bf, Wcp, cproj_b, mp_bf, nullptr, nullptr,
              (int)SR, 64, 256, 0, 0, 0, 1);
  col_attn_k<<<384, 256, 0, stream>>>(mp_bf, colat_f);
  LAUNCH_GEMM(EPI_SIGMUL, false, bf16, mn_bf, Wcg, cgate_b, gated, colat_f, nullptr,
              (int)SR, 64, 256, 0, 0, 0, 1);
  LAUNCH_GEMM(EPI_ADDRES, false, float, gated, Wco, cout_b, m_b, nullptr, m_a,
              (int)SR, 256, 64, 0, 0, 0, 1);

  // ---- Stage C: MSA transition ----
  ln_k<<<(unsigned)(SR / 8), 256, 0, stream>>>(m_b, mtn_g, mtn_b, mn_bf, 256);
  LAUNCH_GEMM(EPI_RELU, false, bf16, mn_bf, Wm1, mt1_b, hid_m, nullptr, nullptr,
              (int)SR, 1024, 256, 0, 0, 0, 1);
  LAUNCH_GEMM(EPI_ADDRES, false, float, hid_m, Wm2, mt2_b, m_a, nullptr, m_b,
              (int)SR, 256, 1024, 0, 0, 0, 1);   // m_a = final m

  // ---- Stage D: outer product mean ----
  ln_k<<<(unsigned)(SR / 8), 256, 0, stream>>>(m_a, on_g, on_b, mn_bf, 256);
  LAUNCH_GEMM(EPI_NONE, false, float, mn_bf, Wpa, opa_b, a_f, nullptr, nullptr,
              (int)SR, 32, 256, 0, 0, 0, 1);
  LAUNCH_GEMM(EPI_NONE, false, float, mn_bf, Wpb, opb_b, b_f, nullptr, nullptr,
              (int)SR, 32, 256, 0, 0, 0, 1);
  transpose_cvt_k<<<dim3(1, (unsigned)(SR / 32)), 256, 0, stream>>>(
      a_f, a_t, (int)SR, 32, 1.0f);
  transpose_cvt_k<<<dim3(1, (unsigned)(SR / 32)), 256, 0, stream>>>(
      b_f, b_t, (int)SR, 32, 1.0f);
  // om[d] = a_t[d]^T @ b_t[d]  (A stored [s x i] = [K x M])
  LAUNCH_GEMM(EPI_NONE, true, float, a_t, b_t, nullptr, om_f, nullptr, nullptr,
              384, 384, 64, SR, SR, RR, 32);
  transpose_cvt_k<<<dim3((unsigned)(RR / 32), 1), 256, 0, stream>>>(
      om_f, om_t, 32, (int)RR, 1.0f / 64.0f);
  LAUNCH_GEMM(EPI_ADDRES, false, float, om_t, Woo, oout_b, z_a, nullptr, z0,
              (int)RR, 128, 32, 0, 0, 0, 1);

  // ---- Stage E: triangle multiplicative update ----
  ln_k<<<(unsigned)(RR / 8), 256, 0, stream>>>(z_a, trn_g, trn_b, zn_bf, 128);
  LAUNCH_GEMM(EPI_SIG, false, float, zn_bf, Wta, tpa_b, ta_f, nullptr, nullptr,
              (int)RR, 64, 128, 0, 0, 0, 1);
  LAUNCH_GEMM(EPI_SIG, false, float, zn_bf, Wtb, tpb_b, tb_f, nullptr, nullptr,
              (int)RR, 64, 128, 0, 0, 0, 1);
  LAUNCH_GEMM(EPI_SIG, false, float, zn_bf, Wtg, tgb, tg_f, nullptr, nullptr,
              (int)RR, 128, 128, 0, 0, 0, 1);
  transpose_cvt_k<<<dim3(2, (unsigned)(RR / 32)), 256, 0, stream>>>(
      ta_f, ta_t, (int)RR, 64, 1.0f);
  transpose_cvt_k<<<dim3(2, (unsigned)(RR / 32)), 256, 0, stream>>>(
      tb_f, tb_t, (int)RR, 64, 1.0f);
  // tu[d] = ta_t[d] ([i][k]) @ tb_t[d] ([k][j])
  LAUNCH_GEMM(EPI_NONE, false, float, ta_t, tb_t, nullptr, tu_f, nullptr, nullptr,
              384, 384, 384, RR, RR, RR, 64);
  ln_tu_k<<<(unsigned)(RR / 256), 256, 0, stream>>>(tu_f, tu_bf, RR);
  LAUNCH_GEMM(EPI_MULADD, false, float, tu_bf, Wto, tout_b, z_b, tg_f, z_a,
              (int)RR, 128, 64, 0, 0, 0, 1);

  // ---- Stage F: pair transition ----
  ln_k<<<(unsigned)(RR / 8), 256, 0, stream>>>(z_b, ptn_g, ptn_b, zn_bf, 128);
  LAUNCH_GEMM(EPI_RELU, false, bf16, zn_bf, Wp1, pt1_b, hid_z, nullptr, nullptr,
              (int)RR, 512, 128, 0, 0, 0, 1);
  LAUNCH_GEMM(EPI_ADDRES, false, float, hid_z, Wp2, pt2_b, z_a, nullptr, z_b,
              (int)RR, 128, 512, 0, 0, 0, 1);    // z_a = final z

  // ---- time mixing + output ----
  time_k<<<1, 64, 0, stream>>>(t_in, tm1_w, tm1_b, tm2_w, tm2_b, mix);
  final_k<<<(unsigned)((MT + ZT + 255) / 256), 256, 0, stream>>>(
      m_a, m0, z_a, z0, mix, (float*)d_out, MT, ZT);
}
